// ViTBackbone_38010460570239
// MI455X (gfx1250) — matrix operand order repaired and checked by reading, not compiled, not run on hardware
//
#include <hip/hip_runtime.h>

// ---------------------------------------------------------------------------
// Types (clang ext vectors only: trivially-copyable, union-safe)
// ---------------------------------------------------------------------------
typedef __attribute__((ext_vector_type(16))) __bf16       v16bf;
typedef __attribute__((ext_vector_type(8)))  float        v8f;
typedef __attribute__((ext_vector_type(4)))  unsigned int v4u;
typedef __attribute__((ext_vector_type(8)))  unsigned int v8u;

union BF16Frag {
    v4u            u4[2];   // two 16-byte chunks
    v16bf          bf;      // WMMA operand
    unsigned short us[16];  // element view
};

#define BB    8
#define NN    1024
#define CC    768
#define HEADS 12
#define HD    64
#define WSZ   32

__device__ __forceinline__ unsigned short f2bf(float f) {
    unsigned int u = __float_as_uint(f);
    u += 0x7FFFu + ((u >> 16) & 1u);          // round-to-nearest-even
    return (unsigned short)(u >> 16);
}
__device__ __forceinline__ float bf2f(unsigned short s) {
    return __uint_as_float(((unsigned int)s) << 16);
}

// ---------------------------------------------------------------------------
// Tensor Data Mover: 2D tile (tile_d1 rows x tile_d0 elems, 2-byte elements)
// global(row-major, stride_d0 elems) -> LDS (packed row-major).
// D# built per CDNA5 ISA; issued via inline asm
// (portable across ROCm7.2 / amdgpu-toolchain builtin arity differences).
// ---------------------------------------------------------------------------
__device__ __forceinline__ void tdm_load_2d_bf16(
    unsigned lds_addr, const void* gptr,
    unsigned tensor_d0, unsigned tensor_d1,
    unsigned tile_d0, unsigned tile_d1, unsigned stride_d0)
{
    unsigned long long ga = (unsigned long long)gptr;
    v4u g0;
    g0[0] = 1u;                                            // count=1, user desc
    g0[1] = lds_addr;                                      // LDS byte address
    g0[2] = (unsigned)ga;                                  // global_addr[31:0]
    g0[3] = (unsigned)((ga >> 32) & 0x1FFFFFFu) | (2u << 30); // [56:32] | type=2
    v8u g1;
    g1[0] = (1u << 16);                                    // data_size=1 (2B)
    g1[1] = (tensor_d0 & 0xFFFFu) << 16;                   // tensor_dim0[15:0]
    g1[2] = (tensor_d0 >> 16) | ((tensor_d1 & 0xFFFFu) << 16);
    g1[3] = (tensor_d1 >> 16) | ((tile_d0 & 0xFFFFu) << 16); // tile_dim0
    g1[4] = (tile_d1 & 0xFFFFu);                           // tile_dim1, tile_dim2=0
    g1[5] = stride_d0;                                     // dim0_stride[31:0]
    g1[6] = 0u;                                            // stride hi, dim1_stride lo
    g1[7] = 0u;
    asm volatile("tensor_load_to_lds %0, %1" :: "s"(g0), "s"(g1) : "memory");
}

// ---------------------------------------------------------------------------
// f32 -> bf16 conversion (weights and activations, one-time)
// ---------------------------------------------------------------------------
__global__ void convert_f32_bf16(const float* __restrict__ in,
                                 unsigned short* __restrict__ out, int n) {
    int i = blockIdx.x * blockDim.x + threadIdx.x;
    if (i < n) out[i] = f2bf(in[i]);
}

// ---------------------------------------------------------------------------
// QKV GEMM: (8192x768 bf16) x (2304x768 bf16)^T -> scatter to q/k/vT bf16.
// q,k head-major (B,heads,N,hd); V stored TRANSPOSED (B,heads,hd,N) so the
// attention P*V B-fragments are contiguous. q pre-scaled by hd^-0.5.
// Weight tile (64 rows x 32 K) staged in LDS by TDM, double-buffered.
// Block: 256 threads (8 waves); wave tile 16(M) x 64(N); K step 32.
// ---------------------------------------------------------------------------
__global__ __launch_bounds__(256) void qkv_gemm(
    const unsigned short* __restrict__ xbf,   // (8192, 768) bf16
    const unsigned short* __restrict__ wbf,   // (2304, 768) bf16
    const float* __restrict__ bias,           // (2304)
    unsigned short* __restrict__ qbf,
    unsigned short* __restrict__ kbf,
    unsigned short* __restrict__ vtbf)        // (B*heads*64, 1024)
{
    const int lane    = threadIdx.x & 31;
    const int wave    = threadIdx.x >> 5;
    const int colBase = blockIdx.x * 64;
    const int rowBase = blockIdx.y * 128 + wave * 16;

    const int am  = lane & 15;          // A-frag row
    const int kb  = (lane >> 4) * 16;   // A-frag K base
    const int bn  = lane & 15;          // B-frag col
    const int kb2 = (lane >> 4) * 16;   // B-frag K base

    __shared__ __align__(16) unsigned short s_w[2][64][32];   // 8KB dbl-buffered

    v8f vz = {};
    v8f acc[4] = {vz, vz, vz, vz};

    const unsigned short* arow = xbf + (size_t)(rowBase + am) * CC;

    // prologue: TDM tile for kk=0 into buffer 0
    if (wave == 0)
        tdm_load_2d_bf16((unsigned)(unsigned long long)(void*)&s_w[0][0][0],
                         wbf + (size_t)colBase * CC, CC, 3 * CC, 32, 64, CC);

    for (int kk = 0; kk < CC; kk += 32) {
        const int buf = (kk >> 5) & 1;
        if (wave == 0) {
            if (kk + 32 < CC) {       // overlap next DMA with this step's math
                tdm_load_2d_bf16(
                    (unsigned)(unsigned long long)(void*)&s_w[buf ^ 1][0][0],
                    wbf + (size_t)colBase * CC + kk + 32, CC, 3 * CC, 32, 64, CC);
                __builtin_amdgcn_s_wait_tensorcnt(1);   // current tile landed
            } else {
                __builtin_amdgcn_s_wait_tensorcnt(0);
            }
        }
        __syncthreads();

        __builtin_prefetch(arow + kk + 64, 0, 1);       // global_prefetch_b8
        BF16Frag a;
        a.u4[0] = *(const v4u*)(arow + kk + kb);
        a.u4[1] = *(const v4u*)(arow + kk + kb + 8);

        BF16Frag b[4];
#pragma unroll
        for (int t = 0; t < 4; ++t) {                   // all B frags first
            b[t].u4[0] = *(const v4u*)(&s_w[buf][t * 16 + bn][kb2]);
            b[t].u4[1] = *(const v4u*)(&s_w[buf][t * 16 + bn][kb2 + 8]);
        }
#pragma unroll
        for (int t = 0; t < 4; ++t)                     // then WMMA burst
            acc[t] = __builtin_amdgcn_wmma_f32_16x16x32_bf16(
                false, a.bf, false, b[t].bf, (short)0, acc[t], false, false);
        __syncthreads();
    }

    // Epilogue: D layout row = i + 8*(lane>=16), col = lane&15
    const int rOff = 8 * (lane >> 4);
    const int dcol = lane & 15;
#pragma unroll
    for (int t = 0; t < 4; ++t) {
        int colG = colBase + t * 16 + dcol;
        int sel  = colG / CC;                 // 0=q 1=k 2=v (uniform per tile)
        int c    = colG - sel * CC;
        int head = c >> 6;
        int d    = c & 63;
        float scl = (sel == 0) ? 0.125f : 1.0f;   // hd^-0.5 = 64^-0.5
        float bi = bias[colG];
#pragma unroll
        for (int i = 0; i < 8; ++i) {
            int rowG = rowBase + rOff + i;
            int bIdx = rowG >> 10;
            int nIdx = rowG & 1023;
            unsigned short v = f2bf((acc[t][i] + bi) * scl);
            if (sel == 2)   // V transposed: [bh][d][key]
                vtbf[((size_t)(bIdx * HEADS + head) * HD + d) * NN + nIdx] = v;
            else {
                unsigned short* dst = (sel == 0) ? qbf : kbf;
                dst[((size_t)(bIdx * HEADS + head) * NN + nIdx) * HD + d] = v;
            }
        }
    }
}

// ---------------------------------------------------------------------------
// Flash attention with decomposed rel-pos bias. 1 wave per (b, head, q-tile16).
// Never materializes the 402MB attn matrix. K-fragments pipelined one
// iteration ahead; V fragments contiguous thanks to transposed V.
// ---------------------------------------------------------------------------
__global__ __launch_bounds__(32) void attn_kernel(
    const unsigned short* __restrict__ qbf,
    const unsigned short* __restrict__ kbf,
    const unsigned short* __restrict__ vtbf,  // (B*heads*64, 1024)
    const float* __restrict__ relph,          // (63, 64)
    const float* __restrict__ relpw,          // (63, 64)
    unsigned short* __restrict__ outbf)       // (8192, 768) bf16
{
    const int lane = threadIdx.x;
    const int qt   = blockIdx.x;              // 0..63
    const int h    = blockIdx.y;              // 0..11
    const int b    = blockIdx.z;              // 0..7
    const int bh   = b * HEADS + h;
    const size_t base  = (size_t)bh * NN * HD;
    const size_t vbase = (size_t)bh * HD * NN;

    __shared__ float s_q[16][HD];
    __shared__ float s_relh[16][WSZ];
    __shared__ float s_relw[16][WSZ];
    __shared__ __align__(16) unsigned short s_p[16][32];

    // Stage scaled Q tile (f32) for bias dot-products
    for (int i = lane; i < 16 * HD; i += 32)
        s_q[i >> 6][i & 63] = bf2f(qbf[base + (size_t)(qt * 16) * HD + i]);
    __syncthreads();

    // rel_h[r][kh] = q_row . rel_pos_h[qy-kh+31], lane = kh (and kw)
#pragma unroll 1
    for (int r = 0; r < 16; ++r) {
        int qidx = qt * 16 + r;
        int qy = qidx >> 5, qx = qidx & 31;
        const float* rh = relph + (size_t)(qy - lane + (WSZ - 1)) * HD;
        const float* rw = relpw + (size_t)(qx - lane + (WSZ - 1)) * HD;
        float ah = 0.f, aw = 0.f;
#pragma unroll
        for (int c = 0; c < HD; ++c) {
            float qv = s_q[r][c];
            ah = fmaf(qv, rh[c], ah);
            aw = fmaf(qv, rw[c], aw);
        }
        s_relh[r][lane] = ah;
        s_relw[r][lane] = aw;
    }
    __syncthreads();

    // Q A-fragments (K split 0..31 / 32..63), contiguous 16B chunks
    const int am  = lane & 15;
    const int kb  = (lane >> 4) * 16;
    const unsigned short* qrow = qbf + base + (size_t)(qt * 16 + am) * HD;
    BF16Frag aq0, aq1;
    aq0.u4[0] = *(const v4u*)(qrow + kb);
    aq0.u4[1] = *(const v4u*)(qrow + kb + 8);
    aq1.u4[0] = *(const v4u*)(qrow + 32 + kb);
    aq1.u4[1] = *(const v4u*)(qrow + 32 + kb + 8);

    const int bn   = lane & 15;
    const int kb2  = (lane >> 4) * 16;
    const int rOff = 8 * (lane >> 4);
    const int dcol = lane & 15;

    float mrow[8], lrow[8];
    v8f vz = {};
    v8f o[4] = {vz, vz, vz, vz};
#pragma unroll
    for (int i = 0; i < 8; ++i) { mrow[i] = -1e30f; lrow[i] = 0.f; }

    // pipelined K fragments (b00,b01 for key tile0; b10,b11 for key tile1)
    BF16Frag bk[4], bkn[4];
    {
        const unsigned short* kr0 = kbf + base + (size_t)(0 + bn) * HD + kb2;
        const unsigned short* kr1 = kbf + base + (size_t)(16 + bn) * HD + kb2;
        bk[0].u4[0] = *(const v4u*)(kr0);      bk[0].u4[1] = *(const v4u*)(kr0 + 8);
        bk[1].u4[0] = *(const v4u*)(kr0 + 32); bk[1].u4[1] = *(const v4u*)(kr0 + 40);
        bk[2].u4[0] = *(const v4u*)(kr1);      bk[2].u4[1] = *(const v4u*)(kr1 + 8);
        bk[3].u4[0] = *(const v4u*)(kr1 + 32); bk[3].u4[1] = *(const v4u*)(kr1 + 40);
    }

    for (int k0 = 0; k0 < NN; k0 += 32) {
        // ---- scores: two 16x16 tiles over hd=64 (4 WMMAs) ----
        v8f s0 = vz, s1 = vz;
        s0 = __builtin_amdgcn_wmma_f32_16x16x32_bf16(false, aq0.bf, false, bk[0].bf, (short)0, s0, false, false);
        s0 = __builtin_amdgcn_wmma_f32_16x16x32_bf16(false, aq1.bf, false, bk[1].bf, (short)0, s0, false, false);
        s1 = __builtin_amdgcn_wmma_f32_16x16x32_bf16(false, aq0.bf, false, bk[2].bf, (short)0, s1, false, false);
        s1 = __builtin_amdgcn_wmma_f32_16x16x32_bf16(false, aq1.bf, false, bk[3].bf, (short)0, s1, false, false);

        // ---- issue loads early: V frags (this iter) + K frags (next iter),
        //      their latency hides under the softmax VALU block ----
        BF16Frag bv[4];
        {
            const unsigned short* vb = vtbf + vbase + (size_t)bn * NN + k0 + kb2;
#pragma unroll
            for (int t = 0; t < 4; ++t) {
                bv[t].u4[0] = *(const v4u*)(vb + (size_t)(t * 16) * NN);
                bv[t].u4[1] = *(const v4u*)(vb + (size_t)(t * 16) * NN + 8);
            }
        }
        if (k0 + 32 < NN) {
            const unsigned short* kr0 = kbf + base + (size_t)(k0 + 32 + bn) * HD + kb2;
            const unsigned short* kr1 = kbf + base + (size_t)(k0 + 48 + bn) * HD + kb2;
            bkn[0].u4[0] = *(const v4u*)(kr0);      bkn[0].u4[1] = *(const v4u*)(kr0 + 8);
            bkn[1].u4[0] = *(const v4u*)(kr0 + 32); bkn[1].u4[1] = *(const v4u*)(kr0 + 40);
            bkn[2].u4[0] = *(const v4u*)(kr1);      bkn[2].u4[1] = *(const v4u*)(kr1 + 8);
            bkn[3].u4[0] = *(const v4u*)(kr1 + 32); bkn[3].u4[1] = *(const v4u*)(kr1 + 40);
        }

        // ---- bias + online softmax (row stats via half-wave shuffles) ----
#pragma unroll
        for (int i = 0; i < 8; ++i) {
            int row = rOff + i;
            int kj0 = k0 + dcol;
            int kj1 = k0 + 16 + dcol;
            float v0 = s0[i] + s_relh[row][kj0 >> 5] + s_relw[row][kj0 & 31];
            float v1 = s1[i] + s_relh[row][kj1 >> 5] + s_relw[row][kj1 & 31];
            float tm = fmaxf(v0, v1);
#pragma unroll
            for (int m = 8; m >= 1; m >>= 1) tm = fmaxf(tm, __shfl_xor(tm, m, 16));
            float mn   = fmaxf(mrow[i], tm);
            float corr = __expf(mrow[i] - mn);
            float p0 = __expf(v0 - mn);
            float p1 = __expf(v1 - mn);
            float rs = p0 + p1;
#pragma unroll
            for (int m = 8; m >= 1; m >>= 1) rs += __shfl_xor(rs, m, 16);
            lrow[i] = lrow[i] * corr + rs;
            mrow[i] = mn;
#pragma unroll
            for (int t = 0; t < 4; ++t) o[t][i] *= corr;
            s_p[row][dcol]      = f2bf(p0);
            s_p[row][16 + dcol] = f2bf(p1);
        }
        __syncthreads();

        // ---- O += P (16x32) x V (32x64), 4 WMMAs ----
        BF16Frag ap;
        ap.u4[0] = *(const v4u*)(&s_p[am][kb]);
        ap.u4[1] = *(const v4u*)(&s_p[am][kb + 8]);
#pragma unroll
        for (int t = 0; t < 4; ++t)
            o[t] = __builtin_amdgcn_wmma_f32_16x16x32_bf16(
                false, ap.bf, false, bv[t].bf, (short)0, o[t], false, false);
        __syncthreads();

#pragma unroll
        for (int t = 0; t < 4; ++t) bk[t] = bkn[t];
    }

    // ---- normalize + store bf16 into (B, N, heads*hd) ----
    const size_t outRowBase = ((size_t)b * NN + qt * 16) * CC + h * HD;
#pragma unroll
    for (int i = 0; i < 8; ++i) {
        float inv = 1.0f / lrow[i];
        int row = rOff + i;
#pragma unroll
        for (int t = 0; t < 4; ++t)
            outbf[outRowBase + (size_t)row * CC + t * 16 + dcol] =
                f2bf(o[t][i] * inv);
    }
}

// ---------------------------------------------------------------------------
// Proj GEMM: (8192x768 bf16) x (768x768 bf16)^T + bias -> f32 d_out.
// Weight tile staged via TDM, double-buffered, same scheme as qkv_gemm.
// ---------------------------------------------------------------------------
__global__ __launch_bounds__(256) void proj_gemm(
    const unsigned short* __restrict__ abf,   // (8192, 768)
    const unsigned short* __restrict__ wbf,   // (768, 768)
    const float* __restrict__ bias,           // (768)
    float* __restrict__ out)                  // (8192, 768)
{
    const int lane    = threadIdx.x & 31;
    const int wave    = threadIdx.x >> 5;
    const int colBase = blockIdx.x * 64;
    const int rowBase = blockIdx.y * 128 + wave * 16;

    const int am  = lane & 15;
    const int kb  = (lane >> 4) * 16;
    const int bn  = lane & 15;
    const int kb2 = (lane >> 4) * 16;

    __shared__ __align__(16) unsigned short s_w[2][64][32];

    v8f vz = {};
    v8f acc[4] = {vz, vz, vz, vz};

    const unsigned short* arow = abf + (size_t)(rowBase + am) * CC;

    if (wave == 0)
        tdm_load_2d_bf16((unsigned)(unsigned long long)(void*)&s_w[0][0][0],
                         wbf + (size_t)colBase * CC, CC, CC, 32, 64, CC);

    for (int kk = 0; kk < CC; kk += 32) {
        const int buf = (kk >> 5) & 1;
        if (wave == 0) {
            if (kk + 32 < CC) {
                tdm_load_2d_bf16(
                    (unsigned)(unsigned long long)(void*)&s_w[buf ^ 1][0][0],
                    wbf + (size_t)colBase * CC + kk + 32, CC, CC, 32, 64, CC);
                __builtin_amdgcn_s_wait_tensorcnt(1);
            } else {
                __builtin_amdgcn_s_wait_tensorcnt(0);
            }
        }
        __syncthreads();

        __builtin_prefetch(arow + kk + 64, 0, 1);
        BF16Frag a;
        a.u4[0] = *(const v4u*)(arow + kk + kb);
        a.u4[1] = *(const v4u*)(arow + kk + kb + 8);

        BF16Frag bf[4];
#pragma unroll
        for (int t = 0; t < 4; ++t) {
            bf[t].u4[0] = *(const v4u*)(&s_w[buf][t * 16 + bn][kb2]);
            bf[t].u4[1] = *(const v4u*)(&s_w[buf][t * 16 + bn][kb2 + 8]);
        }
#pragma unroll
        for (int t = 0; t < 4; ++t)
            acc[t] = __builtin_amdgcn_wmma_f32_16x16x32_bf16(
                false, a.bf, false, bf[t].bf, (short)0, acc[t], false, false);
        __syncthreads();
    }

    const int rOff = 8 * (lane >> 4);
    const int dcol = lane & 15;
#pragma unroll
    for (int t = 0; t < 4; ++t) {
        int colG = colBase + t * 16 + dcol;
        float bi = bias[colG];
#pragma unroll
        for (int i = 0; i < 8; ++i) {
            int rowG = rowBase + rOff + i;
            out[(size_t)rowG * CC + colG] = acc[t][i] + bi;
        }
    }
}

// ---------------------------------------------------------------------------
// Launch
// ---------------------------------------------------------------------------
extern "C" void kernel_launch(void* const* d_in, const int* in_sizes, int n_in,
                              void* d_out, int out_size, void* d_ws, size_t ws_size,
                              hipStream_t stream) {
    const float* x      = (const float*)d_in[0];
    const float* qkv_w  = (const float*)d_in[1];
    const float* qkv_b  = (const float*)d_in[2];
    const float* proj_w = (const float*)d_in[3];
    const float* proj_b = (const float*)d_in[4];
    const float* rel_h  = (const float*)d_in[5];
    const float* rel_w  = (const float*)d_in[6];
    (void)in_sizes; (void)n_in; (void)out_size; (void)ws_size;

    // Workspace layout (ushort elements): ~67 MB total
    unsigned short* ws = (unsigned short*)d_ws;
    size_t off = 0;
    unsigned short* qkvw_bf = ws + off; off += (size_t)3 * CC * CC;          // 2304x768
    unsigned short* projw_bf= ws + off; off += (size_t)CC * CC;              // 768x768
    unsigned short* xbf     = ws + off; off += (size_t)BB * NN * CC;         // 8192x768
    unsigned short* qbf     = ws + off; off += (size_t)BB * HEADS * NN * HD; // head-major
    unsigned short* kbf     = ws + off; off += (size_t)BB * HEADS * NN * HD;
    unsigned short* vtbf    = ws + off; off += (size_t)BB * HEADS * HD * NN; // transposed
    unsigned short* aobf    = ws + off; off += (size_t)BB * NN * CC;         // (8192,768)

    int n1 = 3 * CC * CC;
    convert_f32_bf16<<<(n1 + 255) / 256, 256, 0, stream>>>(qkv_w, qkvw_bf, n1);
    int n2 = CC * CC;
    convert_f32_bf16<<<(n2 + 255) / 256, 256, 0, stream>>>(proj_w, projw_bf, n2);
    int n3 = BB * NN * CC;
    convert_f32_bf16<<<(n3 + 255) / 256, 256, 0, stream>>>(x, xbf, n3);

    qkv_gemm<<<dim3(3 * CC / 64, BB * NN / 128), 256, 0, stream>>>(
        xbf, qkvw_bf, qkv_b, qbf, kbf, vtbf);

    attn_kernel<<<dim3(NN / 16, HEADS, BB), 32, 0, stream>>>(
        qbf, kbf, vtbf, rel_h, rel_w, aobf);

    proj_gemm<<<dim3(CC / 64, BB * NN / 128), 256, 0, stream>>>(
        aobf, projw_bf, proj_b, (float*)d_out);
}
